// CoreAttention_43121471652293
// MI455X (gfx1250) — matrix operand order repaired and checked by reading, not compiled, not run on hardware
//
#include <hip/hip_runtime.h>

typedef __attribute__((ext_vector_type(16))) _Float16 v16h;
typedef __attribute__((ext_vector_type(8)))  _Float16 h8;
typedef __attribute__((ext_vector_type(2)))  _Float16 h2;
typedef __attribute__((ext_vector_type(8)))  float    v8f;
typedef __attribute__((ext_vector_type(4)))  float    f4;
typedef __attribute__((ext_vector_type(2)))  float    f2;

#define SQ 2048
#define BATCH 4
#define NP 16
#define HN 64
#define SCALE 0.125f          // 1/sqrt(64), folded into Q->f16 conversion
#define MASK_FILL -10000.0f
#define QROWS_PER_BLOCK 128   // 8 waves * 16 rows
#define KV_BLK 64
#define PITCH 72              // LDS row pitch in halves: 144B rows, 16B aligned slices

static __device__ __forceinline__ v8f wmma_f16(v16h a, v16h b, v8f c) {
    return __builtin_amdgcn_wmma_f32_16x16x32_f16(
        false, a, false, b, (short)0, c, false, false);
}

static __device__ __forceinline__ v16h cat16(h8 lo, h8 hi) {
    return __builtin_shufflevector(lo, hi, 0,1,2,3,4,5,6,7,8,9,10,11,12,13,14,15);
}

__global__ __launch_bounds__(256)
void flash_attn_fwd(const float* __restrict__ Q,
                    const float* __restrict__ K,
                    const float* __restrict__ V,
                    float* __restrict__ Out) {
    // K row-major [key][hn]; V transposed [hn][key]; per-wave P^T staging [q][key]
    __shared__ __align__(16) _Float16 Ksh[KV_BLK][PITCH];
    __shared__ __align__(16) _Float16 Vts[HN][PITCH];
    __shared__ __align__(16) _Float16 Pt[8][16][PITCH];

    const int tid  = threadIdx.x;
    const int wv   = tid >> 5;
    const int lane = tid & 31;
    const int l16  = lane & 15;
    const int hi   = lane >> 4;            // 0: lanes 0-15, 1: lanes 16-31

    const int b  = blockIdx.y / NP;
    const int h  = blockIdx.y % NP;
    const int q0 = blockIdx.x * QROWS_PER_BLOCK;
    const int qrow_lo = q0 + wv * 16;
    const int qrow    = qrow_lo + l16;     // this lane's query row (S^T: q on lane)

    const size_t bh_off     = (size_t)b * NP * HN + (size_t)h * HN;
    const size_t row_stride = (size_t)BATCH * NP * HN;

    // Cooperative-load index split (fixed per thread across all blocks)
    const int kKey = tid >> 5;             // pass p adds p*8 keys
    const int kHp  = (tid & 31) * 2;       // hn pair base
    const int vHn  = tid & 63;
    const int vKey = (tid >> 6) * 2;       // pass p adds p*8 keys

    // ---- Q^T B-fragments (K-dim = hn): element i <-> hn = c*32 + hi*16 + i ----
    v16h qb0, qb1;
    {
        const float* qp = Q + (size_t)qrow * row_stride + bh_off + hi * 16;
#pragma unroll
        for (int i = 0; i < 16; ++i) {
            qb0[i] = (_Float16)(qp[i]      * SCALE);
            qb1[i] = (_Float16)(qp[32 + i] * SCALE);
        }
    }

    // ---- flash state: lane-local (one q row per lane) ----
    float mrun = -1e30f, lsum = 0.f;
    const v8f vzero = {0.f,0.f,0.f,0.f,0.f,0.f,0.f,0.f};
    v8f oacc[4];                     // O^T tiles: M = hn = t2*16 + hi*8 + r, N = q = l16
#pragma unroll
    for (int t = 0; t < 4; ++t) oacc[t] = vzero;

    const int nkb = (q0 + QROWS_PER_BLOCK - 1) / KV_BLK + 1;

    // ---- prefetch registers: block data staged in VGPRs one iteration ahead ----
    f2 kreg[8], vreg[8];
#pragma unroll
    for (int p = 0; p < 8; ++p) {   // issue loads for block 0
        const float* ks = K + (size_t)(kKey + p * 8) * row_stride + bh_off + kHp;
        kreg[p] = *(const f2*)ks;
        const float* vs = V + (size_t)(vKey + p * 8) * row_stride + bh_off + vHn;
        vreg[p].x = vs[0];
        vreg[p].y = vs[row_stride];
    }

    for (int kb = 0; kb < nkb; ++kb) {
        const int kv0 = kb * KV_BLK;
        __syncthreads();   // previous iteration's LDS reads complete

        // ---- convert + store prefetched block to LDS ----
#pragma unroll
        for (int p = 0; p < 8; ++p) {
            h2 kk = { (_Float16)kreg[p].x, (_Float16)kreg[p].y };
            *(h2*)&Ksh[kKey + p * 8][kHp] = kk;
            h2 vv = { (_Float16)vreg[p].x, (_Float16)vreg[p].y };
            *(h2*)&Vts[vHn][vKey + p * 8] = vv;
        }
        __syncthreads();

        // ---- prefetch next block into registers (overlaps with compute) ----
        if (kb + 1 < nkb) {
            const size_t nb = (size_t)(kv0 + KV_BLK) * row_stride + bh_off;
#pragma unroll
            for (int p = 0; p < 8; ++p) {
                const float* ks = K + nb + (size_t)(kKey + p * 8) * row_stride + kHp;
                kreg[p] = *(const f2*)ks;
                const float* vs = V + nb + (size_t)(vKey + p * 8) * row_stride + vHn;
                vreg[p].x = vs[0];
                vreg[p].y = vs[row_stride];
            }
        }

        // wave-uniform causal skip (scalar branch: EXEC untouched for WMMA)
        if (__builtin_amdgcn_readfirstlane((kv0 <= qrow_lo + 15) ? 1 : 0)) {
            // ---- S^T = K Q^T : 4 key tiles, hn-depth 64 = 2 wmma each ----
            v8f sacc[4];
#pragma unroll
            for (int t = 0; t < 4; ++t) {
                const int krow = t * 16 + l16;
                v16h ka0 = cat16(*(const h8*)&Ksh[krow][hi * 16],
                                 *(const h8*)&Ksh[krow][hi * 16 + 8]);
                v16h ka1 = cat16(*(const h8*)&Ksh[krow][32 + hi * 16],
                                 *(const h8*)&Ksh[krow][32 + hi * 16 + 8]);
                sacc[t] = wmma_f16(ka0, qb0, vzero);
                sacc[t] = wmma_f16(ka1, qb1, sacc[t]);
            }

            // ---- causal mask: only diagonal-straddling blocks (wave-uniform) ----
            if (__builtin_amdgcn_readfirstlane((kv0 + KV_BLK - 1 > qrow_lo) ? 1 : 0)) {
                const int d0 = qrow - kv0 - hi * 8;   // element r of tile t masked iff r > d0 - t*16
#pragma unroll
                for (int t = 0; t < 4; ++t) {
                    const int dt = d0 - t * 16;
#pragma unroll
                    for (int r = 0; r < 8; ++r)
                        if (r > dt) sacc[t][r] = MASK_FILL;
                }
            }

            // ---- online softmax: lane-local + one half-swap shuffle ----
            float mblk = -1e30f;
#pragma unroll
            for (int t = 0; t < 4; ++t)
#pragma unroll
                for (int r = 0; r < 8; ++r) mblk = fmaxf(mblk, sacc[t][r]);
            mblk = fmaxf(mblk, __shfl_xor(mblk, 16, 32));
            const float mnew = fmaxf(mrun, mblk);

            float psum = 0.f;
#pragma unroll
            for (int t = 0; t < 4; ++t) {
                h8 ph;
#pragma unroll
                for (int r = 0; r < 8; ++r) {
                    const float pv = __expf(sacc[t][r] - mnew);
                    psum += pv;
                    ph[r] = (_Float16)pv;
                }
                // P^T row-major: lane owns 8 consecutive keys -> one b128 store
                *(h8*)&Pt[wv][l16][t * 16 + hi * 8] = ph;
            }
            psum += __shfl_xor(psum, 16, 32);
            const float corr = __expf(mrun - mnew);
            lsum = lsum * corr + psum;
            mrun = mnew;
#pragma unroll
            for (int t2 = 0; t2 < 4; ++t2)
#pragma unroll
                for (int r = 0; r < 8; ++r) oacc[t2][r] *= corr;

            // same-wave cross-lane LDS RAW: make P^T stores visible
            asm volatile("s_wait_dscnt 0" ::: "memory");

            // P^T B-frags: elem i <-> key = c*32 + hi*16 + i (contiguous row read)
            v16h pb0 = cat16(*(const h8*)&Pt[wv][l16][hi * 16],
                             *(const h8*)&Pt[wv][l16][hi * 16 + 8]);
            v16h pb1 = cat16(*(const h8*)&Pt[wv][l16][32 + hi * 16],
                             *(const h8*)&Pt[wv][l16][32 + hi * 16 + 8]);

            // ---- O^T += V^T P^T : 4 hn tiles, key-depth 64 = 2 wmma each ----
#pragma unroll
            for (int t2 = 0; t2 < 4; ++t2) {
                const int vrow = t2 * 16 + l16;   // A-frag: lane M = hn
                v16h va0 = cat16(*(const h8*)&Vts[vrow][hi * 16],
                                 *(const h8*)&Vts[vrow][hi * 16 + 8]);
                v16h va1 = cat16(*(const h8*)&Vts[vrow][32 + hi * 16],
                                 *(const h8*)&Vts[vrow][32 + hi * 16 + 8]);
                oacc[t2] = wmma_f16(va0, pb0, oacc[t2]);
                oacc[t2] = wmma_f16(va1, pb1, oacc[t2]);
            }
        }
    }

    // ---- epilogue: lane-local normalize, vectorized stores ----
    const float inv = 1.0f / lsum;
    float* orow = Out + (size_t)qrow * row_stride + bh_off;
#pragma unroll
    for (int t2 = 0; t2 < 4; ++t2) {
        const int hn = t2 * 16 + hi * 8;   // 8 consecutive hn per lane
        f4 o0 = { oacc[t2][0] * inv, oacc[t2][1] * inv,
                  oacc[t2][2] * inv, oacc[t2][3] * inv };
        f4 o1 = { oacc[t2][4] * inv, oacc[t2][5] * inv,
                  oacc[t2][6] * inv, oacc[t2][7] * inv };
        *(f4*)&orow[hn]     = o0;
        *(f4*)&orow[hn + 4] = o1;
    }
}

extern "C" void kernel_launch(void* const* d_in, const int* in_sizes, int n_in,
                              void* d_out, int out_size, void* d_ws, size_t ws_size,
                              hipStream_t stream) {
    const float* Q = (const float*)d_in[0];
    const float* K = (const float*)d_in[1];
    const float* V = (const float*)d_in[2];
    // d_in[3] = attention_mask (bool): causal mask reproduced analytically.
    float* Out = (float*)d_out;

    dim3 grid(SQ / QROWS_PER_BLOCK, BATCH * NP);   // (16, 64)
    dim3 block(256);                                // 8 wave32
    flash_attn_fwd<<<grid, block, 0, stream>>>(Q, K, V, Out);
}
